// Autoencoder_87205015978715
// MI455X (gfx1250) — matrix operand order repaired and checked by reading, not compiled, not run on hardware
//
#include <hip/hip_runtime.h>
#include <hip/hip_bf16.h>
#include <math.h>

// ---------------------------------------------------------------------------
// Problem constants (from reference)
// ---------------------------------------------------------------------------
#define B_SZ     16384
#define D_IN     2048
#define HIDDEN   1536
#define N_COMP   256
#define TOP_K    4

typedef __bf16 bhalf;
typedef __attribute__((ext_vector_type(16))) __bf16 v16bf;
typedef __attribute__((ext_vector_type(8)))  float  v8f;

__device__ __forceinline__ float gelu_exact(float x) {
    return 0.5f * x * (1.0f + erff(x * 0.70710678118654752440f));
}

// CDNA5 async global->LDS copy (16B per lane), tracked by ASYNCcnt.
// lds_off: per-lane LDS byte offset; voff: per-lane 32-bit global byte offset;
// base: uniform 64-bit base address (SGPR pair) -> GVS addressing mode.
__device__ __forceinline__ void async_copy_b128(unsigned lds_off, unsigned voff,
                                                const void* base) {
    asm volatile("global_load_async_to_lds_b128 %0, %1, %2 offset:0"
                 :: "v"(lds_off), "v"(voff), "s"(base) : "memory");
}
__device__ __forceinline__ void wait_async0() {
    asm volatile("s_wait_asynccnt 0x0" ::: "memory");
}

// ---------------------------------------------------------------------------
// Kernel: broadcast components (256x3) to encoded output (B,256,3)
// grid = (3, B), block = 256
// ---------------------------------------------------------------------------
__global__ __launch_bounds__(256) void k_encoded_bcast(
    const float* __restrict__ comps, float* __restrict__ out)
{
    int j = blockIdx.x * 256 + threadIdx.x;          // 0..767
    size_t row = blockIdx.y;
    out[row * 768 + j] = comps[j];
}

// ---------------------------------------------------------------------------
// Kernel: xb = bf16(x + embed[layer_idx])   grid = (D_IN/256, B)
// ---------------------------------------------------------------------------
__global__ __launch_bounds__(256) void k_convert_x(
    const float* __restrict__ x, const int* __restrict__ layer_idx,
    const float* __restrict__ embed, bhalf* __restrict__ xb)
{
    int col = blockIdx.x * 256 + threadIdx.x;        // 0..2047
    size_t row = blockIdx.y;
    int li = *layer_idx;
    float v = x[row * D_IN + col] + embed[(size_t)li * D_IN + col];
    xb[row * D_IN + col] = (bhalf)v;
}

// ---------------------------------------------------------------------------
// Kernel: 32x32 tiled transpose + fp32->bf16:  in (K,N) -> out (N,K)
// grid = (K/32, N/32), block = 256
// ---------------------------------------------------------------------------
__global__ __launch_bounds__(256) void k_transpose_bf16(
    const float* __restrict__ in, bhalf* __restrict__ out, int K, int N)
{
    __shared__ float tile[32][33];
    int k0 = blockIdx.x * 32, n0 = blockIdx.y * 32;
    int tx = threadIdx.x & 31, ty = threadIdx.x >> 5;   // ty: 0..7
    #pragma unroll
    for (int i = ty; i < 32; i += 8)
        tile[i][tx] = in[(size_t)(k0 + i) * N + n0 + tx];
    __syncthreads();
    #pragma unroll
    for (int i = ty; i < 32; i += 8)
        out[(size_t)(n0 + i) * K + k0 + tx] = (bhalf)tile[tx][i];
}

// ---------------------------------------------------------------------------
// WMMA GEMM:  C[M,N] = act( A[M,K](bf16,row) * Bt[N,K](bf16,row)^T + bias )
// Block tile 128x128x64, 256 threads = 8 waves, wave tile 64x32 (4x2 WMMA).
// Double-buffered LDS fed by global_load_async_to_lds_b128 (no VGPR staging,
// no ds_store, no spills); fragment reads are ds_load_b128.
// ---------------------------------------------------------------------------
#define BM 128
#define BN 128
#define BK 64
#define KPAD 72                      // 144B row stride: 16B aligned, de-conflicted
#define TILE_BYTES (BM * KPAD * 2)   // 18432 B per matrix per buffer

template<bool GELU, bool OUTBF16>
__global__ __launch_bounds__(256) void k_gemm_bf16(
    const bhalf* __restrict__ A, const bhalf* __restrict__ Bt,
    const float* __restrict__ bias, void* __restrict__ Cout,
    int M, int N, int K)
{
    __shared__ __align__(16) bhalf lA[2][BM * KPAD];
    __shared__ __align__(16) bhalf lB[2][BN * KPAD];

    const int tid   = threadIdx.x;
    const int wave  = tid >> 5;
    const int lane  = tid & 31;
    const int bm    = blockIdx.y * BM;
    const int bn    = blockIdx.x * BN;
    const int waveM = (wave & 1) * 64;      // 2 waves along M
    const int waveN = (wave >> 1) * 32;     // 4 waves along N
    const int nk    = K / BK;

    v8f acc[4][2];
    #pragma unroll
    for (int i = 0; i < 4; ++i)
        #pragma unroll
        for (int j = 0; j < 2; ++j)
            acc[i][j] = (v8f){0.f,0.f,0.f,0.f,0.f,0.f,0.f,0.f};

    // Per-thread tile-slot mapping: 1024 uint4 slots per 128x64 tile,
    // 4 slots per thread per matrix.  Loop-invariant byte offsets.
    unsigned ldsA_off[4], ldsB_off[4], gA_off[4], gB_off[4];
    const unsigned lAbase = (unsigned)(unsigned long long)(void*)&lA[0][0];
    const unsigned lBbase = (unsigned)(unsigned long long)(void*)&lB[0][0];
    #pragma unroll
    for (int u = 0; u < 4; ++u) {
        int e = tid + u * 256;              // 0..1023
        int row = e >> 3, c8 = (e & 7) * 8; // 128 rows x 8 uint4
        ldsA_off[u] = lAbase + (unsigned)((row * KPAD + c8) * 2);
        ldsB_off[u] = lBbase + (unsigned)((row * KPAD + c8) * 2);
        gA_off[u]   = (unsigned)(((size_t)(bm + row) * K + c8) * 2);
        gB_off[u]   = (unsigned)(((size_t)(bn + row) * K + c8) * 2);
    }

    auto issueAsync = [&](int kt, int buf) {
        const unsigned kofs = (unsigned)(kt * BK * 2);
        const unsigned bofs = (unsigned)(buf * TILE_BYTES);
        #pragma unroll
        for (int u = 0; u < 4; ++u)
            async_copy_b128(ldsA_off[u] + bofs, gA_off[u] + kofs, (const void*)A);
        #pragma unroll
        for (int u = 0; u < 4; ++u)
            async_copy_b128(ldsB_off[u] + bofs, gB_off[u] + kofs, (const void*)Bt);
    };

    auto compute = [&](int buf) {
        const int rsel  = lane & 15;
        const int khalf = lane >> 4;
        #pragma unroll
        for (int ks = 0; ks < 2; ++ks) {
            const int kk = ks * 32;
            v16bf afrag[4], bfrag[2];
            #pragma unroll
            for (int mt = 0; mt < 4; ++mt) {
                const bhalf* p = &lA[buf][(waveM + mt * 16 + rsel) * KPAD];
                union { v16bf v; uint4 q[2]; } u;
                u.q[0] = *(const uint4*)(p + kk + khalf * 8);       // K: kk+kh*8 .. +7
                u.q[1] = *(const uint4*)(p + kk + 16 + khalf * 8);  // K: kk+16+kh*8 ..
                afrag[mt] = u.v;
            }
            #pragma unroll
            for (int nt = 0; nt < 2; ++nt) {
                const bhalf* p = &lB[buf][(waveN + nt * 16 + rsel) * KPAD];
                union { v16bf v; uint4 q[2]; } u;
                u.q[0] = *(const uint4*)(p + kk + khalf * 8);       // K: kk+kh*8 .. +7
                u.q[1] = *(const uint4*)(p + kk + 16 + khalf * 8);  // K: kk+16+kh*8 ..
                bfrag[nt] = u.v;
            }
            #pragma unroll
            for (int mt = 0; mt < 4; ++mt)
                #pragma unroll
                for (int nt = 0; nt < 2; ++nt)
                    acc[mt][nt] = __builtin_amdgcn_wmma_f32_16x16x32_bf16(
                        false, afrag[mt], false, bfrag[nt],
                        (short)0, acc[mt][nt], false, false);
        }
    };

    // Prologue: fill buffer 0
    issueAsync(0, 0);
    wait_async0();
    __syncthreads();

    for (int kt = 0; kt < nk; ++kt) {
        const int cur = kt & 1;
        if (kt + 1 < nk) issueAsync(kt + 1, cur ^ 1);   // async fill of other buffer
        compute(cur);                                    // WMMAs hide copy latency
        wait_async0();                                   // my async copies landed
        __syncthreads();                                 // everyone's copies visible
    }

    // Epilogue: C/D layout -> row = vgpr + 8*(lane>>4), col = lane&15
    const int lrow = 8 * (lane >> 4);
    const int lcol = lane & 15;
    #pragma unroll
    for (int mt = 0; mt < 4; ++mt) {
        #pragma unroll
        for (int nt = 0; nt < 2; ++nt) {
            const int c = bn + waveN + nt * 16 + lcol;
            const float bv = bias[c];
            #pragma unroll
            for (int i = 0; i < 8; ++i) {
                const int r = bm + waveM + mt * 16 + lrow + i;
                float v = acc[mt][nt][i] + bv;
                if (GELU) v = gelu_exact(v);
                if (OUTBF16) ((bhalf*)Cout)[(size_t)r * N + c] = (bhalf)v;
                else         ((float*)Cout)[(size_t)r * N + c] = v;
            }
        }
    }
}

// ---------------------------------------------------------------------------
// Kernel: top-4 + softmax + component mix. One wave32 per row, 8 rows/block.
// grid = B/8, block = 256
// ---------------------------------------------------------------------------
__global__ __launch_bounds__(256) void k_topk_mix(
    const float* __restrict__ logits, const float* __restrict__ comps,
    float* __restrict__ pts)
{
    const int wave = threadIdx.x >> 5;
    const int lane = threadIdx.x & 31;
    const size_t row = (size_t)blockIdx.x * 8 + wave;

    float v[8];
    #pragma unroll
    for (int j = 0; j < 8; ++j)
        v[j] = logits[row * N_COMP + lane + 32 * j];

    float topv[TOP_K];
    int   topi[TOP_K];
    #pragma unroll
    for (int t = 0; t < TOP_K; ++t) {
        float lm = -3.0e38f; int lj = 0;
        #pragma unroll
        for (int j = 0; j < 8; ++j)
            if (v[j] > lm) { lm = v[j]; lj = j; }
        float m  = lm;
        int   gi = lane + 32 * lj;
        #pragma unroll
        for (int off = 16; off >= 1; off >>= 1) {
            float om = __shfl_xor(m, off, 32);
            int   oi = __shfl_xor(gi, off, 32);
            if (om > m || (om == m && oi < gi)) { m = om; gi = oi; }
        }
        topv[t] = m; topi[t] = gi;
        if ((gi & 31) == lane) v[gi >> 5] = -3.0e38f;   // retire winner
    }

    if (lane == 0) {
        float mx = topv[0];
        float e[TOP_K], s = 0.f;
        #pragma unroll
        for (int t = 0; t < TOP_K; ++t) { e[t] = __expf(topv[t] - mx); s += e[t]; }
        float inv = 1.0f / s;
        float p0 = 0.f, p1 = 0.f, p2 = 0.f;
        #pragma unroll
        for (int t = 0; t < TOP_K; ++t) {
            float w = e[t] * inv;
            const float* c = comps + 3 * topi[t];
            p0 += w * c[0]; p1 += w * c[1]; p2 += w * c[2];
        }
        pts[row * 3 + 0] = p0;
        pts[row * 3 + 1] = p1;
        pts[row * 3 + 2] = p2;
    }
}

// ---------------------------------------------------------------------------
// Kernel: d = bf16(gelu(pts @ dec_w1 + dec_b1)).  grid = (HIDDEN/256, B)
// ---------------------------------------------------------------------------
__global__ __launch_bounds__(256) void k_dec1(
    const float* __restrict__ pts, const float* __restrict__ w,
    const float* __restrict__ b, bhalf* __restrict__ out)
{
    int j = blockIdx.x * 256 + threadIdx.x;      // 0..1535
    size_t row = blockIdx.y;
    float p0 = pts[row * 3 + 0], p1 = pts[row * 3 + 1], p2 = pts[row * 3 + 2];
    float v = p0 * w[j] + p1 * w[HIDDEN + j] + p2 * w[2 * HIDDEN + j] + b[j];
    out[row * HIDDEN + j] = (bhalf)gelu_exact(v);
}

// ---------------------------------------------------------------------------
// Host-side orchestration
// ---------------------------------------------------------------------------
extern "C" void kernel_launch(void* const* d_in, const int* in_sizes, int n_in,
                              void* d_out, int out_size, void* d_ws, size_t ws_size,
                              hipStream_t stream)
{
    (void)in_sizes; (void)n_in; (void)out_size; (void)ws_size;

    const float* x       = (const float*)d_in[0];
    const int*   lidx    = (const int*)  d_in[1];
    const float* embed   = (const float*)d_in[2];
    const float* enc_w1  = (const float*)d_in[3];
    const float* enc_b1  = (const float*)d_in[4];
    const float* enc_w2  = (const float*)d_in[5];
    const float* enc_b2  = (const float*)d_in[6];
    const float* comps   = (const float*)d_in[7];
    const float* dec_w1  = (const float*)d_in[8];
    const float* dec_b1  = (const float*)d_in[9];
    const float* dec_w2  = (const float*)d_in[10];
    const float* dec_b2  = (const float*)d_in[11];

    float* out_encoded = (float*)d_out;
    float* out_decoded = (float*)d_out + (size_t)B_SZ * N_COMP * 3;

    // workspace layout (bytes, all 256-aligned). logits & db reuse dead xb.
    char* ws = (char*)d_ws;
    bhalf* xb     = (bhalf*)(ws + 0);                       // 67,108,864 B
    float* logits = (float*)(ws + 0);                       // 16,777,216 B (after GEMM1)
    bhalf* db     = (bhalf*)(ws + 16777216);                // 50,331,648 B (after topk)
    bhalf* w1b    = (bhalf*)(ws + 67108864);                //  6,291,456 B
    bhalf* w2b    = (bhalf*)(ws + 73400320);                //    786,432 B
    bhalf* w4b    = (bhalf*)(ws + 74186752);                //  6,291,456 B
    bhalf* hb     = (bhalf*)(ws + 80478208);                // 50,331,648 B
    float* pts    = (float*)(ws + 130809856);               //    196,608 B

    // 1) encoded output broadcast
    k_encoded_bcast<<<dim3(3, B_SZ), 256, 0, stream>>>(comps, out_encoded);

    // 2) xb = bf16(x + embed[layer_idx])
    k_convert_x<<<dim3(D_IN / 256, B_SZ), 256, 0, stream>>>(x, lidx, embed, xb);

    // 3) weight transposes (K,N)->(N,K) + bf16
    k_transpose_bf16<<<dim3(D_IN / 32, HIDDEN / 32), 256, 0, stream>>>(enc_w1, w1b, D_IN, HIDDEN);
    k_transpose_bf16<<<dim3(HIDDEN / 32, N_COMP / 32), 256, 0, stream>>>(enc_w2, w2b, HIDDEN, N_COMP);
    k_transpose_bf16<<<dim3(HIDDEN / 32, D_IN / 32), 256, 0, stream>>>(dec_w2, w4b, HIDDEN, D_IN);

    // 4) GEMM1: hb = gelu(xb @ enc_w1 + b1)   (bf16 out)
    k_gemm_bf16<true, true><<<dim3(HIDDEN / BN, B_SZ / BM), 256, 0, stream>>>(
        xb, w1b, enc_b1, (void*)hb, B_SZ, HIDDEN, D_IN);

    // 5) GEMM2: logits = hb @ enc_w2 + b2     (f32 out)
    k_gemm_bf16<false, false><<<dim3(N_COMP / BN, B_SZ / BM), 256, 0, stream>>>(
        hb, w2b, enc_b2, (void*)logits, B_SZ, N_COMP, HIDDEN);

    // 6) top-4 + softmax + component mix -> pts (B,3)
    k_topk_mix<<<B_SZ / 8, 256, 0, stream>>>(logits, comps, pts);

    // 7) db = bf16(gelu(pts @ dec_w1 + b1))
    k_dec1<<<dim3(HIDDEN / 256, B_SZ), 256, 0, stream>>>(pts, dec_w1, dec_b1, db);

    // 8) GEMM4: decoded = db @ dec_w2 + b2    (f32 out to d_out)
    k_gemm_bf16<false, false><<<dim3(D_IN / BN, B_SZ / BM), 256, 0, stream>>>(
        db, w4b, dec_b2, (void*)out_decoded, B_SZ, D_IN, HIDDEN);
}
